// MultiHeadSelfAttention2D_51951924412912
// MI455X (gfx1250) — matrix operand order repaired and checked by reading, not compiled, not run on hardware
//
#include <hip/hip_runtime.h>
#include <math.h>

// CDNA5 / gfx1250: wave32, WMMA 16x16x32 bf16 (D = A*B + C, f32 accum)
typedef __attribute__((ext_vector_type(16))) __bf16 v16bf;
typedef __attribute__((ext_vector_type(8)))  float  v8f;
typedef unsigned int v4u __attribute__((ext_vector_type(4)));
typedef int          v4i __attribute__((ext_vector_type(4)));
typedef int          v8i __attribute__((ext_vector_type(8)));

namespace {
constexpr int kCIn  = 256;
constexpr int kCOut = 512;
constexpr int kHeads = 8;
constexpr int kDh   = 64;   // kCOut / kHeads
constexpr int kB    = 16;
constexpr int kM    = 1024; // 32*32
}

// Tensor Data Mover available? (device pass only; host pass takes fallback)
#if defined(__HIP_DEVICE_COMPILE__) && __has_builtin(__builtin_amdgcn_tensor_load_to_lds)
#define USE_TDM 1
#else
#define USE_TDM 0
#endif

static __device__ __forceinline__ v8f wmma_bf16(v16bf a, v16bf b, v8f c) {
  // (neg_a, A, neg_b, B, c_mod, C, reuse_a, reuse_b)
  return __builtin_amdgcn_wmma_f32_16x16x32_bf16(false, a, false, b, (short)0, c,
                                                 false, false);
}

// 16-bit A-matrix (16x32) per-lane K index for element j (ISA 7.12.2):
// lanes 0-15: VGPR0-3 -> K=0..7, VGPR4-7 -> K=16..23 ; lanes 16-31: +8
static __device__ __forceinline__ int a_kmap(int j, int hi) {
  return (j & 7) | ((j >> 3) << 4) | (hi << 3);
}

// Sinusoidal 2D positional encoding value for (channel c, position m).
static __device__ __forceinline__ float pe_val(int c, int m) {
  int pos, cc;
  if (c < kCOut / 2) { pos = m >> 5; cc = c; }
  else               { pos = m & 31; cc = c - kCOut / 2; }
  float two_p = (float)(cc & ~1);                       // 2 * (cc/2)
  float div = __expf(-9.210340371976184f * two_p * (1.0f / 256.0f)); // ln(1e4)
  float ang = (float)pos * div;
  return (cc & 1) ? __cosf(ang) : __sinf(ang);
}

// Issue a TDM 1-D copy of `bytes` (multiple of 8) from global to LDS.
// D# per ISA 08_async_tensor §8.3/8.4: 1-D tile of 8-byte elements.
static __device__ __forceinline__ void tdm_load_1d(const void* gsrc, void* ldst,
                                                   unsigned bytes) {
#if USE_TDM
  const unsigned n8 = bytes >> 3;                 // elements of 8 bytes
  const unsigned long long ga = (unsigned long long)(uintptr_t)gsrc;
  const unsigned la = (unsigned)(uintptr_t)ldst;  // LDS aperture: low 32 bits

  v4u g0;
  g0[0] = 1u;                                        // count=1, user mode
  g0[1] = la;                                        // lds_addr
  g0[2] = (unsigned)ga;                              // global_addr[31:0]
  g0[3] = (unsigned)((ga >> 32) & 0x1FFFFFFu) | 0x80000000u; // [56:32] | type=2

  v8i g1;
  g1[0] = 0x00030000;              // workgroup_mask=0, data_size=3 (8B)
  g1[1] = (int)(n8 << 16);         // tensor_dim0[15:0] in bits 63:48
  g1[2] = (int)(1u << 16);         // tensor_dim0[31:16]=0, tensor_dim1=1
  g1[3] = (int)(n8 << 16);         // tile_dim0 in bits 127:112
  g1[4] = 1;                       // tile_dim1=1, tile_dim2=0
  g1[5] = (int)n8;                 // tensor_dim0_stride[31:0]
  g1[6] = (int)(n8 << 16);         // tensor_dim1_stride[15:0]
  g1[7] = 0;

  v4i gz = {0, 0, 0, 0};
#if __has_include(<hip/amd_detail/amd_gfx1250_TDM.h>)
  v8i gz8 = {0, 0, 0, 0, 0, 0, 0, 0};
  __builtin_amdgcn_tensor_load_to_lds(g0, g1, gz, gz, gz8, 0);   // clang-23 form
#else
  __builtin_amdgcn_tensor_load_to_lds(g0, g1, gz, gz, 0);        // ROCm 7.2 form
#endif
#else
  (void)gsrc; (void)ldst; (void)bytes;
#endif
}

// s_wait_tensorcnt needs a constant immediate -> template non-type parameter.
template <short N>
static __device__ __forceinline__ void wait_tensorcnt() {
#if USE_TDM
  __builtin_amdgcn_s_wait_tensorcnt(N);
#endif
}

// ---------------------------------------------------------------------------
// Kernel 1: Q/K/V = W{q,k,v} @ x per batch (1x1 conv as channel GEMM), PE added
// to Q and K, results stored bf16 in (b, head, m, d) layout.
// One wave computes one 16(row) x 16(m) tile for all three matrices.
// ---------------------------------------------------------------------------
__global__ __launch_bounds__(128) void qkv_pe_kernel(
    const float* __restrict__ x,   // (B, C_IN, M)
    const float* __restrict__ Wq,  // (C_OUT, C_IN)
    const float* __restrict__ Wk,
    const float* __restrict__ Wv,
    __bf16* __restrict__ Qb,       // (B, HEADS, M, DH) bf16, PE added
    __bf16* __restrict__ Kb,
    __bf16* __restrict__ Vb) {
  const int lane = threadIdx.x & 31;
  const int waveId = blockIdx.x * (blockDim.x >> 5) + (threadIdx.x >> 5);
  const int mt = waveId & 63;          // 64 m-tiles of 16
  const int rt = (waveId >> 6) & 31;   // 32 row-tiles of 16
  const int b  = waveId >> 11;         // 16 batches
  const int hi = lane >> 4;
  const int ln = lane & 15;

  const int row  = rt * 16 + ln;       // A fragment: M = lane&15 (both halves)
  const int mcol = mt * 16 + ln;       // B fragment: N = lane&15 (both halves)

  v8f aq = {}, ak = {}, av = {};
  for (int kk = 0; kk < kCIn / 32; ++kk) {
    v16bf fa_q, fa_k, fa_v;
#pragma unroll
    for (int j = 0; j < 16; ++j) {
      const int cin = kk * 32 + a_kmap(j, hi);
      fa_q[j] = (__bf16)Wq[(size_t)row * kCIn + cin];
      fa_k[j] = (__bf16)Wk[(size_t)row * kCIn + cin];
      fa_v[j] = (__bf16)Wv[(size_t)row * kCIn + cin];
    }
    // B fragment (32 x 16): lane holds column mcol, K(cin) = a_kmap(j, hi)
    v16bf fb;
#pragma unroll
    for (int j = 0; j < 16; ++j) {
      const int cin = kk * 32 + a_kmap(j, hi);
      fb[j] = (__bf16)x[((size_t)b * kCIn + cin) * kM + mcol];
    }
    aq = wmma_bf16(fa_q, fb, aq);
    ak = wmma_bf16(fa_k, fb, ak);
    av = wmma_bf16(fa_v, fb, av);
  }

  // D layout: VGPR r -> out-channel row r + 8*hi, lane&15 -> column m.
#pragma unroll
  for (int r = 0; r < 8; ++r) {
    const int c = rt * 16 + r + (hi << 3);
    const float pe = pe_val(c, mcol);
    const int h = c >> 6, d = c & 63;
    const size_t idx = (((size_t)b * kHeads + h) * kM + mcol) * kDh + d;
    Qb[idx] = (__bf16)(aq[r] + pe);
    Kb[idx] = (__bf16)(ak[r] + pe);
    Vb[idx] = (__bf16)(av[r]);
  }
}

// ---------------------------------------------------------------------------
// Kernel 2: flash-style attention. One block (4 waves) owns 4 consecutive
// 16-query tiles of a single (b, head). K/V key-chunks (32 keys x 64 d bf16
// = 4KB each, contiguous in memory) are staged into LDS once per block:
// double-buffered TDM tensor_load_to_lds when available, cooperative
// synchronous copy otherwise.
// ---------------------------------------------------------------------------
__global__ __launch_bounds__(128) void attn_kernel(
    const __bf16* __restrict__ Qb, const __bf16* __restrict__ Kb,
    const __bf16* __restrict__ Vb, float* __restrict__ out) {
  __shared__ __bf16 ldsK[2][32 * kDh];  // 2 x 4KB
  __shared__ __bf16 ldsV[2][32 * kDh];  // 2 x 4KB
  __shared__ __bf16 plds[4][16 * 32];   // per-wave P staging tile

  const int tid  = threadIdx.x;
  const int lane = tid & 31;
  const int wib  = tid >> 5;
  const int qt = (blockIdx.x & 15) * 4 + wib;  // 4 query tiles per block
  const int h  = (blockIdx.x >> 4) & 7;
  const int b  = blockIdx.x >> 7;
  const int hi = lane >> 4;
  const int ln = lane & 15;

  const size_t hb = ((size_t)b * kHeads + h) * kM;
  const __bf16* Qh = Qb + hb * kDh;
  const __bf16* Kh = Kb + hb * kDh;
  const __bf16* Vh = Vb + hb * kDh;

  // Q A-fragments (16 queries x 64 d as two 16x32 fragments); row = query.
  v16bf qa0, qa1;
  const int q0 = qt * 16 + ln;
#pragma unroll
  for (int j = 0; j < 16; ++j) {
    const int K = j + (hi << 4);
    qa0[j] = Qh[(size_t)q0 * kDh + K];
    qa1[j] = Qh[(size_t)q0 * kDh + 32 + K];
  }

  float mrow[8], lrow[8];
  v8f o0 = {}, o1 = {}, o2 = {}, o3 = {};
#pragma unroll
  for (int r = 0; r < 8; ++r) { mrow[r] = -1e30f; lrow[r] = 0.0f; }

#if USE_TDM
  // Prologue: DMA chunk 0 into buffer 0 (wave 0 issues; EXEC-independent).
  if (wib == 0) {
    tdm_load_1d(Kh, &ldsK[0][0], 32 * kDh * 2);
    tdm_load_1d(Vh, &ldsV[0][0], 32 * kDh * 2);
  }
#endif

  for (int nc = 0; nc < kM / 32; ++nc) {
    const int kbase = nc * 32;
#if USE_TDM
    const int cur = nc & 1;
    if (wib == 0) {
      if (nc + 1 < kM / 32) {
        const int nxt = (nc + 1) & 1;
        tdm_load_1d(Kh + (size_t)(kbase + 32) * kDh, &ldsK[nxt][0], 32 * kDh * 2);
        tdm_load_1d(Vh + (size_t)(kbase + 32) * kDh, &ldsV[nxt][0], 32 * kDh * 2);
        wait_tensorcnt<2>();  // current chunk's 2 DMAs complete
      } else {
        wait_tensorcnt<0>();
      }
    }
    __syncthreads();  // current buffers valid for all waves
#else
    const int cur = 0;
    {  // cooperative synchronous stage: 128 threads x 32B per matrix
      const uint4* gk = (const uint4*)(Kh + (size_t)kbase * kDh);
      const uint4* gv = (const uint4*)(Vh + (size_t)kbase * kDh);
      uint4* lk = (uint4*)&ldsK[0][0];
      uint4* lv = (uint4*)&ldsV[0][0];
      lk[tid] = gk[tid];
      lk[tid + 128] = gk[tid + 128];
      lv[tid] = gv[tid];
      lv[tid + 128] = gv[tid + 128];
    }
    __syncthreads();
#endif
    const __bf16* Kc = &ldsK[cur][0];
    const __bf16* Vc = &ldsV[cur][0];

    // K B-fragments from LDS: S(16q x 16n) = Qa(16x32 d) x B(32 d x 16 n),
    // B[d][n] = K[n][d]; lane column n, K(d) = j + 16*hi (contiguous per lane).
    const v16bf kb0a = *(const v16bf*)(Kc + (size_t)ln * kDh + (hi << 4));
    const v16bf kb0b = *(const v16bf*)(Kc + (size_t)ln * kDh + 32 + (hi << 4));
    const v16bf kb1a = *(const v16bf*)(Kc + (size_t)(16 + ln) * kDh + (hi << 4));
    const v16bf kb1b = *(const v16bf*)(Kc + (size_t)(16 + ln) * kDh + 32 + (hi << 4));

    v8f s0 = {}, s1 = {};
    s0 = wmma_bf16(qa0, kb0a, s0);
    s0 = wmma_bf16(qa1, kb0b, s0);
    s1 = wmma_bf16(qa0, kb1a, s1);
    s1 = wmma_bf16(qa1, kb1b, s1);

#pragma unroll
    for (int r = 0; r < 8; ++r) {
      s0[r] = fminf(fmaxf(s0[r] * 0.125f, -50.0f), 50.0f);
      s1[r] = fminf(fmaxf(s1[r] * 0.125f, -50.0f), 50.0f);
    }

    // Online softmax. Row (query) = r + 8*hi lives across the 16 lanes of a
    // half-wave; butterfly over masks 1,2,4,8 stays within the half.
#pragma unroll
    for (int r = 0; r < 8; ++r) {
      float v = fmaxf(s0[r], s1[r]);
#pragma unroll
      for (int mask = 1; mask <= 8; mask <<= 1)
        v = fmaxf(v, __shfl_xor(v, mask, 32));
      const float mnew = fmaxf(mrow[r], v);
      const float alpha = __expf(mrow[r] - mnew);
      mrow[r] = mnew;

      const float p0 = __expf(s0[r] - mnew);
      const float p1 = __expf(s1[r] - mnew);
      float ps = p0 + p1;
#pragma unroll
      for (int mask = 1; mask <= 8; mask <<= 1)
        ps += __shfl_xor(ps, mask, 32);
      lrow[r] = lrow[r] * alpha + ps;

      o0[r] *= alpha; o1[r] *= alpha; o2[r] *= alpha; o3[r] *= alpha;

      // Stage P (16 x 32) to LDS in row-major for A-fragment repack.
      plds[wib][(r + (hi << 3)) * 32 + ln]      = (__bf16)p0;
      plds[wib][(r + (hi << 3)) * 32 + 16 + ln] = (__bf16)p1;
    }

    // Repack P into 16-bit A-fragment layout (row = lane&15, K = key-in-chunk).
    v16bf pa;
#pragma unroll
    for (int j = 0; j < 16; ++j) pa[j] = plds[wib][ln * 32 + a_kmap(j, hi)];

    // O(16q x 16d) += P(16x32 key) x V^T(32 key x 16 d); B[key][d] = Vc[key*DH+d]
#pragma unroll
    for (int dt = 0; dt < 4; ++dt) {
      v16bf vbf;
#pragma unroll
      for (int j = 0; j < 16; ++j) {
        const int key = a_kmap(j, hi);
        vbf[j] = Vc[(size_t)key * kDh + dt * 16 + ln];
      }
      if (dt == 0) o0 = wmma_bf16(pa, vbf, o0);
      else if (dt == 1) o1 = wmma_bf16(pa, vbf, o1);
      else if (dt == 2) o2 = wmma_bf16(pa, vbf, o2);
      else o3 = wmma_bf16(pa, vbf, o3);
    }
    __syncthreads();  // all waves done with current buffers before overwrite
  }

  // Normalize by row sums and scatter to out (B, C_OUT, M); col(lane) = d.
#pragma unroll
  for (int r = 0; r < 8; ++r) {
    const float inv = 1.0f / lrow[r];
    const int q = qt * 16 + r + (hi << 3);
    const size_t base = ((size_t)b * kCOut + h * kDh) * kM + q;
    out[base + (size_t)(0 * 16 + ln) * kM] = o0[r] * inv;
    out[base + (size_t)(1 * 16 + ln) * kM] = o1[r] * inv;
    out[base + (size_t)(2 * 16 + ln) * kM] = o2[r] * inv;
    out[base + (size_t)(3 * 16 + ln) * kM] = o3[r] * inv;
  }
}

// ---------------------------------------------------------------------------
extern "C" void kernel_launch(void* const* d_in, const int* in_sizes, int n_in,
                              void* d_out, int out_size, void* d_ws,
                              size_t ws_size, hipStream_t stream) {
  const float* x  = (const float*)d_in[0];
  const float* Wq = (const float*)d_in[1];
  const float* Wk = (const float*)d_in[2];
  const float* Wv = (const float*)d_in[3];
  float* out = (float*)d_out;

  const size_t nQKV = (size_t)kB * kHeads * kM * kDh;  // 8.4M elems each
  __bf16* Qb = (__bf16*)d_ws;
  __bf16* Kb = Qb + nQKV;
  __bf16* Vb = Kb + nQKV;  // total 48 MB of workspace (bf16)

  // Projection + PE: 16 batches * 32 row-tiles * 64 m-tiles = 32768 waves.
  qkv_pe_kernel<<<8192, 128, 0, stream>>>(x, Wq, Wk, Wv, Qb, Kb, Vb);
  // Attention: 2048 blocks, each = one (b, head) x 4 query tiles.
  attn_kernel<<<2048, 128, 0, stream>>>(Qb, Kb, Vb, out);
}
